// CNN_Selective_Attention_59201829208690
// MI455X (gfx1250) — matrix operand order repaired and checked by reading, not compiled, not run on hardware
//
#include <hip/hip_runtime.h>
#include <hip/hip_bf16.h>

// ---------------------------------------------------------------------------
// PCNN + selective attention for MI455X (gfx1250, wave32, WMMA).
//
// Problem sizes (from reference):
//   B=1024 sentences, L=128 tokens, DIM=50, DWPE=5, WIN=3, C=230, REL=53
//   Lout = L - WIN + 1 = 126
//
// Conv-as-GEMM: X[l, k] (k = j*64 + d, j=window pos, d=embedding dim, padded
// 60->64) times WpadT (filters, pre-transposed + zero padded to [240][192]).
// Embeddings staged in LDS at row stride 64 bf16 -> X row l is the 192
// contiguous bf16 at l*64; A-fragments come straight out of LDS.
// Staging uses v_cvt_pk_bf16_f32 (1 VALU op per bf16 pair) and float2 loads
// of wordVec rows (rows are 200B -> 8B aligned).
// ---------------------------------------------------------------------------

#define WORDS     1000000
#define BAG       1024
#define SLEN      128
#define DIMW      50
#define DWPE      5
#define WINSZ     3
#define CFLT      230
#define RELT      53
#define LOUT      126     // SLEN - WINSZ + 1

#define DSTRIDE   64      // padded per-position embedding width (bf16)
#define KPAD      192     // WINSZ * DSTRIDE
#define CPAD      240     // CFLT padded to multiple of 16
#define KTILES    6       // KPAD / 32
#define NTILES    15      // CPAD / 16
#define MTILES    8       // SLEN / 16 (last tile partially masked)
#define LROWS     130     // SLEN + (WINSZ-1) zero-pad rows in LDS

// output layout in d_out (floats): output | att_output | cnn_output | weight
#define OFF_OUT   0
#define OFF_ATT   (RELT)                       // 53
#define OFF_CNN   (RELT + CFLT)                // 283
#define OFF_WGT   (RELT + CFLT + BAG * CFLT)   // 235803

typedef __attribute__((ext_vector_type(16))) __bf16 v16bf;
typedef __attribute__((ext_vector_type(8)))  __bf16 v8bf;
typedef __attribute__((ext_vector_type(2)))  __bf16 v2bf;
typedef __attribute__((ext_vector_type(8)))  float  v8f;

__device__ __forceinline__ unsigned int f32_to_bf16_bits(float f) {
    union { float f; unsigned int u; } x;
    x.f = f;
    unsigned int u = x.u;
    u += 0x7FFFu + ((u >> 16) & 1u);   // round-to-nearest-even
    return u >> 16;
}

// Two f32 -> packed bf16x2 in a single VALU op where the HW converter exists.
__device__ __forceinline__ unsigned int pack_bf16x2(float lo, float hi) {
#if __has_builtin(__builtin_amdgcn_cvt_pk_bf16_f32)
    union { v2bf v; unsigned int u; } c;
    c.v = __builtin_amdgcn_cvt_pk_bf16_f32(lo, hi);
    return c.u;
#else
    return f32_to_bf16_bits(lo) | (f32_to_bf16_bits(hi) << 16);
#endif
}

// ---------------------------------------------------------------------------
// Kernel 1: build bf16 transposed/padded filter bank WpadT[CPAD][KPAD] and
// fused bias bsum[CPAD].  Wv:[3,50,230]  Wp1/Wp2:[3,5,230] (row major).
// ---------------------------------------------------------------------------
__global__ __launch_bounds__(256)
void prep_weights_kernel(const float* __restrict__ Wv,
                         const float* __restrict__ Wp1,
                         const float* __restrict__ Wp2,
                         const float* __restrict__ bv,
                         const float* __restrict__ bp1,
                         const float* __restrict__ bp2,
                         unsigned short* __restrict__ wpadT,
                         float* __restrict__ bsum) {
    const int idx = blockIdx.x * blockDim.x + threadIdx.x;
    if (idx < CPAD * KPAD) {
        const int n = idx / KPAD;
        const int k = idx % KPAD;
        const int j = k / DSTRIDE;
        const int d = k % DSTRIDE;
        float val = 0.0f;
        if (n < CFLT) {
            if (d < DIMW)               val = Wv [(j * DIMW + d) * CFLT + n];
            else if (d < DIMW + DWPE)   val = Wp1[(j * DWPE + (d - DIMW)) * CFLT + n];
            else if (d < DIMW + 2*DWPE) val = Wp2[(j * DWPE + (d - DIMW - DWPE)) * CFLT + n];
        }
        wpadT[n * KPAD + k] = (unsigned short)f32_to_bf16_bits(val);
    }
    if (idx < CPAD)
        bsum[idx] = (idx < CFLT) ? (bv[idx] + bp1[idx] + bp2[idx]) : 0.0f;
}

// ---------------------------------------------------------------------------
// Kernel 2: attv[c1] = sum_c2 attW[c1,c2] * MR[c2, rel]   (collapses the
// attention bilinear form to one vector; only column `rel` is ever used).
// ---------------------------------------------------------------------------
__global__ __launch_bounds__(256)
void attv_kernel(const float* __restrict__ attW,
                 const float* __restrict__ MR,
                 const int*   __restrict__ relp,
                 float* __restrict__ attv) {
    const int c1 = blockIdx.x * blockDim.x + threadIdx.x;
    if (c1 < CFLT) {
        const int rel = relp[0];
        float s = 0.0f;
        for (int c2 = 0; c2 < CFLT; ++c2)
            s += attW[c1 * CFLT + c2] * MR[c2 * RELT + rel];
        attv[c1] = s;
    }
}

// ---------------------------------------------------------------------------
// Kernel 3: per-sentence PCNN encoder.  One workgroup (4 wave32) per bag
// element: gather embeddings -> bf16 LDS (packed u32 stores), WMMA GEMM,
// fused max/bias/tanh epilogue.
// ---------------------------------------------------------------------------
__global__ __launch_bounds__(128)
void conv_pool_kernel(const int*   __restrict__ sen,
                      const int*   __restrict__ p1,
                      const int*   __restrict__ p2,
                      const float* __restrict__ wordVec,
                      const float* __restrict__ posVec1,
                      const float* __restrict__ posVec2,
                      const unsigned short* __restrict__ wpadT,
                      const float* __restrict__ bsum,
                      float* __restrict__ cnn) {
    __shared__ unsigned short ldsEmb[LROWS * DSTRIDE];   // 16.6 KB

    const int b   = blockIdx.x;
    const int tid = threadIdx.x;

    // ---- stage embeddings: thread t converts token row t (128 rows) -------
    {
        const int l = tid;
        unsigned int* row32 =
            reinterpret_cast<unsigned int*>(&ldsEmb[l * DSTRIDE]); // 128B align
        const int wi = sen[b * SLEN + l];
        // wordVec row = 200B -> 8B aligned -> float2 loads are safe
        const float2* wp =
            reinterpret_cast<const float2*>(wordVec + (size_t)wi * DIMW);
        #pragma unroll
        for (int d = 0; d < DIMW / 2; ++d) {      // 25 packed stores
            const float2 v = wp[d];
            row32[d] = pack_bf16x2(v.x, v.y);
        }
        // position embeddings: 10 floats -> 5 packed stores (cols 50..59)
        const int i1 = p1[b * SLEN + l];
        const int i2 = p2[b * SLEN + l];
        const float* q1 = posVec1 + i1 * DWPE;
        const float* q2 = posVec2 + i2 * DWPE;
        float tmp[2 * DWPE];
        #pragma unroll
        for (int d = 0; d < DWPE; ++d) { tmp[d] = q1[d]; tmp[DWPE + d] = q2[d]; }
        #pragma unroll
        for (int d = 0; d < DWPE; ++d)            // pairs (50,51)...(58,59)
            row32[DIMW / 2 + d] = pack_bf16x2(tmp[2 * d], tmp[2 * d + 1]);
        row32[30] = 0u;                           // pad cols 60..63
        row32[31] = 0u;
        if (tid < LROWS - SLEN) {                 // zero-pad rows 128,129
            unsigned int* pr =
                reinterpret_cast<unsigned int*>(&ldsEmb[(SLEN + tid) * DSTRIDE]);
            #pragma unroll
            for (int d = 0; d < DSTRIDE / 2; ++d) pr[d] = 0u;
        }
    }
    __syncthreads();

    const int wave = tid >> 5;
    const int lane = tid & 31;
    const int r    = lane & 15;      // row-in-tile / column-in-tile
    const int half = lane >> 4;      // lane half; selects K 16*half..16*half+15 in this code

    for (int nt = wave; nt < NTILES; nt += 4) {
        const int n = nt * 16 + r;   // filter index owned by this lane pair
        // B fragments: lane holds column n, 16 consecutive K (per half).
        v16bf bf[KTILES];
        #pragma unroll
        for (int kt = 0; kt < KTILES; ++kt)
            bf[kt] = *reinterpret_cast<const v16bf*>(
                wpadT + (size_t)n * KPAD + kt * 32 + half * 16);

        float cmax = -3.402823466e38f;
        for (int mt = 0; mt < MTILES; ++mt) {
            v8f acc = {};
            const int l = mt * 16 + r;           // sentence position (row M=r)
            #pragma unroll
            for (int kt = 0; kt < KTILES; ++kt) {
                // A fragment: elems 0-15 = K(kt*32 + half*16 ..), contiguous.
                union { v16bf v; v8bf h[2]; } a;
                const unsigned short* base =
                    &ldsEmb[l * DSTRIDE + kt * 32 + half * 16];
                a.h[0] = *reinterpret_cast<const v8bf*>(base);
                a.h[1] = *reinterpret_cast<const v8bf*>(base + 8);
                acc = __builtin_amdgcn_wmma_f32_16x16x32_bf16(
                    false, a.v, false, bf[kt], (short)0, acc, false, false);
            }
            // max over the 8 M rows this lane holds; last tile masks l>=126
            // (lanes half==1 hold M=8..15 -> l=120..127, drop the last two).
            const int nvalid = (mt == MTILES - 1) ? (half ? 6 : 8) : 8;
            #pragma unroll
            for (int i = 0; i < 8; ++i)
                if (i < nvalid) cmax = fmaxf(cmax, acc[i]);
        }
        // merge the two M-halves of column n (lanes j <-> j+16)
        const float other = __shfl_xor(cmax, 16, 32);
        cmax = fmaxf(cmax, other);
        if (half == 0 && n < CFLT)
            cnn[(size_t)b * CFLT + n] = tanhf(cmax + bsum[n]);
    }
}

// ---------------------------------------------------------------------------
// Kernel 4: s[b] = dot(cnn[b,:], attv) + MRb[rel]
// ---------------------------------------------------------------------------
__global__ __launch_bounds__(256)
void score_kernel(const float* __restrict__ cnn,
                  const float* __restrict__ attv,
                  const float* __restrict__ MRb,
                  const int*   __restrict__ relp,
                  float* __restrict__ s) {
    __shared__ float red[256];
    const int b = blockIdx.x, t = threadIdx.x;
    float p = 0.0f;
    if (t < CFLT) p = cnn[(size_t)b * CFLT + t] * attv[t];
    red[t] = p;
    __syncthreads();
    for (int off = 128; off > 0; off >>= 1) {
        if (t < off) red[t] += red[t + off];
        __syncthreads();
    }
    if (t == 0) s[b] = red[0] + MRb[relp[0]];
}

// ---------------------------------------------------------------------------
// Kernel 5: weight = softmax over the bag of s[0..1023]
// ---------------------------------------------------------------------------
__global__ __launch_bounds__(1024)
void softmax_weight_kernel(const float* __restrict__ s,
                           float* __restrict__ weight) {
    __shared__ float red[1024];
    const int t = threadIdx.x;
    const float x = s[t];
    red[t] = x;
    __syncthreads();
    for (int off = 512; off > 0; off >>= 1) {
        if (t < off) red[t] = fmaxf(red[t], red[t + off]);
        __syncthreads();
    }
    const float m = red[0];
    __syncthreads();
    const float e = expf(x - m);
    red[t] = e;
    __syncthreads();
    for (int off = 512; off > 0; off >>= 1) {
        if (t < off) red[t] += red[t + off];
        __syncthreads();
    }
    weight[t] = e / red[0];
}

// ---------------------------------------------------------------------------
// Kernel 6: att[c] = sum_b weight[b] * cnn[b, c]
// ---------------------------------------------------------------------------
__global__ __launch_bounds__(256)
void att_kernel(const float* __restrict__ weight,
                const float* __restrict__ cnn,
                float* __restrict__ att) {
    __shared__ float red[256];
    const int c = blockIdx.x, t = threadIdx.x;
    float p = 0.0f;
    for (int b = t; b < BAG; b += 256)
        p += weight[b] * cnn[(size_t)b * CFLT + c];
    red[t] = p;
    __syncthreads();
    for (int off = 128; off > 0; off >>= 1) {
        if (t < off) red[t] += red[t + off];
        __syncthreads();
    }
    if (t == 0) att[c] = red[0];
}

// ---------------------------------------------------------------------------
// Kernel 7: output = softmax(att @ MR + MRb)   (1x53)
// ---------------------------------------------------------------------------
__global__ __launch_bounds__(64)
void final_kernel(const float* __restrict__ att,
                  const float* __restrict__ MR,
                  const float* __restrict__ MRb,
                  float* __restrict__ out) {
    __shared__ float red[64];
    const int t = threadIdx.x;
    float logit = -3.402823466e38f;
    if (t < RELT) {
        float s = MRb[t];
        for (int c = 0; c < CFLT; ++c) s += att[c] * MR[c * RELT + t];
        logit = s;
    }
    red[t] = logit;
    __syncthreads();
    for (int off = 32; off > 0; off >>= 1) {
        if (t < off) red[t] = fmaxf(red[t], red[t + off]);
        __syncthreads();
    }
    const float m = red[0];
    __syncthreads();
    const float e = (t < RELT) ? expf(logit - m) : 0.0f;
    red[t] = e;
    __syncthreads();
    for (int off = 32; off > 0; off >>= 1) {
        if (t < off) red[t] += red[t + off];
        __syncthreads();
    }
    if (t < RELT) out[t] = e / red[0];
}

// ---------------------------------------------------------------------------
extern "C" void kernel_launch(void* const* d_in, const int* in_sizes, int n_in,
                              void* d_out, int out_size, void* d_ws, size_t ws_size,
                              hipStream_t stream) {
    const int*   sen     = (const int*)  d_in[0];
    const int*   p1      = (const int*)  d_in[1];
    const int*   p2      = (const int*)  d_in[2];
    const float* wordVec = (const float*)d_in[3];
    const float* posVec1 = (const float*)d_in[4];
    const float* posVec2 = (const float*)d_in[5];
    const float* Wv      = (const float*)d_in[6];
    const float* Wp1     = (const float*)d_in[7];
    const float* Wp2     = (const float*)d_in[8];
    const float* bv      = (const float*)d_in[9];
    const float* bp1     = (const float*)d_in[10];
    const float* bp2     = (const float*)d_in[11];
    const float* attW    = (const float*)d_in[12];
    const float* MR      = (const float*)d_in[13];
    const float* MRb     = (const float*)d_in[14];
    const int*   relp    = (const int*)  d_in[15];

    float* out = (float*)d_out;
    float* att = out + OFF_ATT;
    float* cnn = out + OFF_CNN;
    float* wgt = out + OFF_WGT;

    // workspace carve-out
    unsigned short* wpadT = (unsigned short*)d_ws;                    // 92160 B
    float* bsum = (float*)((char*)d_ws + (size_t)CPAD * KPAD * 2);    //   960 B
    float* attv = bsum + CPAD;                                        //   960 B
    float* sarr = attv + CPAD;                                        //  4096 B

    prep_weights_kernel<<<(CPAD * KPAD + 255) / 256, 256, 0, stream>>>(
        Wv, Wp1, Wp2, bv, bp1, bp2, wpadT, bsum);
    attv_kernel<<<1, 256, 0, stream>>>(attW, MR, relp, attv);
    conv_pool_kernel<<<BAG, 128, 0, stream>>>(
        sen, p1, p2, wordVec, posVec1, posVec2, wpadT, bsum, cnn);
    score_kernel<<<BAG, 256, 0, stream>>>(cnn, attv, MRb, relp, sarr);
    softmax_weight_kernel<<<1, 1024, 0, stream>>>(sarr, wgt);
    att_kernel<<<CFLT, 256, 0, stream>>>(wgt, cnn, att);
    final_kernel<<<1, 64, 0, stream>>>(att, MR, MRb, out);
}
